// MFNet_78434692759804
// MI455X (gfx1250) — matrix operand order repaired and checked by reading, not compiled, not run on hardware
//
#include <hip/hip_runtime.h>

#define NUM_USER   4096
#define NUM_ITEM   16384
#define NUM_HIDDEN 64
#define BATCH      1024
#define SETLEN     50

typedef __attribute__((ext_vector_type(16))) _Float16 v16h;
typedef __attribute__((ext_vector_type(8)))  float    v8f;
typedef __attribute__((ext_vector_type(4)))  float    f4;

// ---------------------------------------------------------------------------
// Kernel A: zero-fill both dense output planes (536 MB) with 128-bit
// non-temporal stores (global_store_b128, TH=NT) and reset the unique counter.
// This is the bandwidth-dominant phase: ~23 us at 23.3 TB/s.
// ---------------------------------------------------------------------------
__global__ void mf_fill_zero(float* __restrict__ out, size_t n4,
                             unsigned* __restrict__ counter) {
    size_t gid    = (size_t)blockIdx.x * blockDim.x + threadIdx.x;
    size_t stride = (size_t)gridDim.x * blockDim.x;
    f4 z = {0.0f, 0.0f, 0.0f, 0.0f};
    f4* p = (f4*)out;
    for (size_t i = gid; i < n4; i += stride) {
        __builtin_nontemporal_store(z, p + i);
    }
    if (gid == 0) *counter = 0u;
}

// ---------------------------------------------------------------------------
// Kernel B: one wave handles 16 (user,item) pairs. A-row m <- embed_user[u_m],
// B-col m <- embed_item[i_m], gathered with one contiguous k order (k = 16h + s) in the A(16x32)/B(32x16)
// lane layouts so the k-ordering matches between A and B. Two
// v_wmma_f32_16x16x32_f16 ops (K=0..31, K=32..63) accumulate in f32; the
// diagonal D[m][m] is the desired dot product. Diag(m) lives at:
//   m in 0..7  -> lane m,     C-vgpr m
//   m in 8..15 -> lane 16+m,  C-vgpr m-8
// Those 16 "owner" lanes scatter pred, label, and the uniqueness atomic.
// ---------------------------------------------------------------------------
__global__ void mf_scatter_wmma(const int*   __restrict__ idx_user,
                                const int*   __restrict__ item_sets,
                                const float* __restrict__ rating_sets,
                                const float* __restrict__ embed_user,
                                const float* __restrict__ embed_item,
                                float*       __restrict__ out_pred,
                                float*       __restrict__ out_label,
                                unsigned*    __restrict__ counter) {
    const int lane = threadIdx.x & 31;
    const int wave = blockIdx.x * (blockDim.x >> 5) + (threadIdx.x >> 5);
    const int m    = lane & 15;
    const bool hi  = lane >= 16;

    const int p  = wave * 16 + m;          // pair index in [0, B*L)
    const int b  = p / SETLEN;
    const int u  = idx_user[b];
    const int it = item_sets[p];           // (B,L) flat == p
    const float rating = rating_sets[p];

    const float* urow = embed_user + (size_t)u  * NUM_HIDDEN;
    const float* irow = embed_item + (size_t)it * NUM_HIDDEN;

    // A 16x32 f16, contiguous k order used here: lane m   holds K = {0..15}
    //                     lane m+16 holds K = {16..31}
    // B 32x16 f16, same contiguous order used here: lane m   holds K = {0..15}
    //                     lane m+16 holds K = {16..31}
    const int a0 = hi ? 16 : 0;
    const int a1 = hi ? 24 : 8;
    const int b0 = hi ? 16 : 0;

    v8f c = {};
#pragma unroll
    for (int kk = 0; kk < NUM_HIDDEN; kk += 32) {
        v16h av, bv;
#pragma unroll
        for (int j = 0; j < 8; ++j)  av[j]     = (_Float16)urow[kk + a0 + j];
#pragma unroll
        for (int j = 0; j < 8; ++j)  av[8 + j] = (_Float16)urow[kk + a1 + j];
#pragma unroll
        for (int j = 0; j < 16; ++j) bv[j]     = (_Float16)irow[kk + b0 + j];
        // (neg_a, A, neg_b, B, c_mod, C, reuse_a, reuse_b)
        c = __builtin_amdgcn_wmma_f32_16x16x32_f16(
                false, av, false, bv, (short)0, c, false, false);
    }

    // Extract diagonal element owned by this lane (lanes 0..7 and 24..31).
    const bool owner = (lane < 8) || (lane >= 24);
    const int  e     = (lane < 8) ? lane : (lane - 24);
    float diag = c[0];
#pragma unroll
    for (int r = 1; r < 8; ++r) diag = (e == r) ? c[r] : diag;

    if (owner) {
        size_t pos = (size_t)u * NUM_ITEM + (size_t)it;
        out_pred[pos] = diag;   // mask == 1.0 at scatter points
        // label scatter + exact unique-position count: label plane was zeroed,
        // so the first writer to a position observes bit-pattern 0.
        unsigned old = atomicExch((unsigned*)&out_label[pos],
                                  __float_as_uint(rating));
        if (old == 0u) atomicAdd(counter, 1u);
    }
}

// ---------------------------------------------------------------------------
// Kernel C: sparsity = NUM_USER*NUM_ITEM / sum(mask) (sum == unique count).
// ---------------------------------------------------------------------------
__global__ void mf_finalize(const unsigned* __restrict__ counter,
                            float* __restrict__ out_sparsity) {
    unsigned cnt = *counter;
    if (cnt == 0u) cnt = 1u;
    *out_sparsity = (float)((double)NUM_USER * (double)NUM_ITEM / (double)cnt);
}

extern "C" void kernel_launch(void* const* d_in, const int* in_sizes, int n_in,
                              void* d_out, int out_size, void* d_ws, size_t ws_size,
                              hipStream_t stream) {
    const int*   idx_user    = (const int*)  d_in[0];
    const int*   item_sets   = (const int*)  d_in[1];
    const float* rating_sets = (const float*)d_in[2];
    const float* embed_user  = (const float*)d_in[3];
    const float* embed_item  = (const float*)d_in[4];

    float* out          = (float*)d_out;
    float* out_pred     = out;
    float* out_label    = out + (size_t)NUM_USER * NUM_ITEM;
    float* out_sparsity = out + 2ull * (size_t)NUM_USER * NUM_ITEM;
    unsigned* counter   = (unsigned*)d_ws;

    // Phase 1: stream zeros over both 256 MiB planes (exactly divisible by 4).
    size_t n4 = (2ull * (size_t)NUM_USER * NUM_ITEM) / 4;
    mf_fill_zero<<<2048, 256, 0, stream>>>(out, n4, counter);

    // Phase 2: 51200 pairs / 16 per wave = 3200 waves; 8 waves per block.
    const int waves = (BATCH * SETLEN) / 16;       // 3200
    mf_scatter_wmma<<<waves / 8, 256, 0, stream>>>(
        idx_user, item_sets, rating_sets, embed_user, embed_item,
        out_pred, out_label, counter);

    // Phase 3: scalar sparsity.
    mf_finalize<<<1, 1, 0, stream>>>(counter, out_sparsity);
}
